// Attention_21045339750416
// MI455X (gfx1250) — matrix operand order repaired and checked by reading, not compiled, not run on hardware
//
#include <hip/hip_runtime.h>
#include <stdint.h>

typedef __attribute__((ext_vector_type(16))) _Float16 v16h;
typedef __attribute__((ext_vector_type(8)))  _Float16 v8h;
typedef __attribute__((ext_vector_type(4)))  _Float16 v4h;
typedef __attribute__((ext_vector_type(8)))  float    v8f;
typedef __attribute__((ext_vector_type(4)))  float    v4f;

constexpr int BATCH = 2;
constexpr int SEQ   = 2048;
constexpr int CH    = 512;
constexpr int HEADS = 8;
constexpr int HD    = 64;   // head dim

constexpr int KBLK  = 64;   // keys staged per iteration
constexpr int QW    = 16;   // q rows per wave
constexpr int WAVES = 4;    // waves per workgroup
constexpr int QB    = QW * WAVES; // 64 q rows per workgroup
constexpr int LSTR  = 72;   // LDS row stride in halves (144B = 9*16B: 16B-aligned rows, conflict padding)

// ============================================================================
// Prepass: convert K -> f16 row-major per head [bh][key][64],
//          convert V -> f16 transposed per head [bh][dim][SEQ]
// Runs once per launch; output lives in d_ws (8 MB total -> L2 resident).
// ============================================================================
__launch_bounds__(256)
__global__ void convert_kv(const float* __restrict__ K, const float* __restrict__ V,
                           _Float16* __restrict__ Kh, _Float16* __restrict__ Vth) {
  __shared__ _Float16 tile[64 * LSTR];   // V block [key][dim], padded
  const int bh = blockIdx.y, b = bh / HEADS, h = bh % HEADS;
  const int k0 = blockIdx.x * 64;
  const int tid = threadIdx.x;
  // 64 rows x 16 float4-chunks = 1024 chunks; 256 threads x 4 iters (coalesced)
  for (int i = 0; i < 4; ++i) {
    const int id  = tid + 256 * i;
    const int row = id >> 4;        // key within block
    const int c   = id & 15;        // float4 chunk within 64 dims
    v4f f = *(const v4f*)(K + ((size_t)(b * SEQ + k0 + row) * CH + h * HD + c * 4));
    v4h hh; for (int u = 0; u < 4; ++u) hh[u] = (_Float16)f[u];
    *(v4h*)&Kh[((size_t)bh * SEQ + k0 + row) * HD + c * 4] = hh;
    v4f g = *(const v4f*)(V + ((size_t)(b * SEQ + k0 + row) * CH + h * HD + c * 4));
    v4h gh; for (int u = 0; u < 4; ++u) gh[u] = (_Float16)g[u];
    *(v4h*)&tile[row * LSTR + c * 4] = gh;
  }
  __syncthreads();
  // transpose write: 64 dims x 16 key-chunks of 4 keys (coalesced along keys)
  for (int i = 0; i < 4; ++i) {
    const int id = tid + 256 * i;
    const int d  = id >> 4;
    const int kc = id & 15;
    v4h hh;
    for (int u = 0; u < 4; ++u) hh[u] = tile[(kc * 4 + u) * LSTR + d];
    *(v4h*)&Vth[((size_t)bh * HD + d) * SEQ + k0 + kc * 4] = hh;
  }
}

// ============================================================================
// Path A: flash attention with async global->LDS staging of pre-converted f16
// K/V tiles (double-buffered LDS, ASYNCcnt-tracked DMA under the WMMA work).
// ============================================================================
__launch_bounds__(WAVES * 32)
__global__ void fa_wmma_async(const float* __restrict__ Q,
                              const _Float16* __restrict__ Kh,
                              const _Float16* __restrict__ Vth,
                              float* __restrict__ O) {
  __shared__ _Float16 Kl[2][KBLK * LSTR];       // [key][dim], f16, double-buffered
  __shared__ _Float16 Vl[2][HD * LSTR];         // [dim][key], f16, double-buffered
  __shared__ _Float16 Plds[WAVES][QW * LSTR];   // per-wave P scratch, [q][key]

  const int bh   = blockIdx.y;
  const int b    = bh / HEADS;
  const int h    = bh % HEADS;
  const int q0   = blockIdx.x * QB;
  const int tid  = threadIdx.x;
  const int wave = tid >> 5;
  const int lane = tid & 31;
  const int l16  = lane & 15;
  const int half = lane >> 4;

  const float sl = 0.125f * 1.44269504088896f;  // D^-0.5 * log2(e)

  // ---- Q^T as B-fragments: lane = query col (l16), element e <-> dim = 32c + 16*half + e
  const int qrow = q0 + wave * QW + l16;
  const float* qptr = Q + ((size_t)(b * SEQ + qrow) * CH + h * HD);
  v16h qb[2];
  for (int c = 0; c < 2; ++c) {
    const float* p0 = qptr + 32 * c + 16 * half;
    v4f f0 = *(const v4f*)(p0);
    v4f f1 = *(const v4f*)(p0 + 4);
    v4f f2 = *(const v4f*)(p0 + 8);
    v4f f3 = *(const v4f*)(p0 + 12);
    for (int i = 0; i < 4; ++i) {
      qb[c][i]      = (_Float16)(f0[i] * sl);
      qb[c][4 + i]  = (_Float16)(f1[i] * sl);
      qb[c][8 + i]  = (_Float16)(f2[i] * sl);
      qb[c][12 + i] = (_Float16)(f3[i] * sl);
    }
  }

  float m = -1e30f, l = 0.f;
  v8f vzero = {};
  v8f acc[4];
  for (int t = 0; t < 4; ++t) acc[t] = vzero;

  const _Float16* khb = Kh + (size_t)bh * SEQ * HD;   // [key][64]
  const _Float16* vtb = Vth + (size_t)bh * HD * SEQ;  // [dim][SEQ]

  // per-thread async staging: 4 x 16B chunks each for K and V per block
  auto stage = [&](int kb, int bufi) {
    #pragma unroll
    for (int i = 0; i < 4; ++i) {
      const int id  = tid + 128 * i;
      const int row = id >> 3;     // K: key row / V: dim row
      const int c   = id & 7;      // 16B chunk (8 halves)
      const _Float16* gk = khb + ((size_t)(kb + row) * HD + c * 8);
      unsigned lk = (unsigned)(uintptr_t)&Kl[bufi][row * LSTR + c * 8];
      asm volatile("global_load_async_to_lds_b128 %0, %1, off"
                   :: "v"(lk), "v"(gk) : "memory");
      const _Float16* gv = vtb + ((size_t)row * SEQ + kb + c * 8);
      unsigned lv = (unsigned)(uintptr_t)&Vl[bufi][row * LSTR + c * 8];
      asm volatile("global_load_async_to_lds_b128 %0, %1, off"
                   :: "v"(lv), "v"(gv) : "memory");
    }
  };

  stage(0, 0);                      // prologue: DMA block 0 into buffer 0

  for (int kb = 0, it = 0; kb < SEQ; kb += KBLK, ++it) {
    const int cur = it & 1;
    asm volatile("s_wait_asynccnt 0" ::: "memory");   // this wave's DMA done
    __syncthreads();                                  // all waves' DMA visible
    if (kb + KBLK < SEQ) stage(kb + KBLK, cur ^ 1);   // next block flies under compute

    const _Float16* Kc = &Kl[cur][0];
    const _Float16* Vc = &Vl[cur][0];

    // ---------- S^T = K * Q^T : 4 tiles (16 keys x 16 queries) ----------
    v8f s[4];
    for (int t = 0; t < 4; ++t) {
      v8f sa = vzero;
      const int keyr = t * 16 + l16;
      for (int c = 0; c < 2; ++c) {
        v8h lo = *(const v8h*)&Kc[keyr * LSTR + 32 * c + 16 * half];
        v8h hi = *(const v8h*)&Kc[keyr * LSTR + 32 * c + 16 * half + 8];
        v16h ka;
        for (int i = 0; i < 8; ++i) { ka[i] = lo[i]; ka[8 + i] = hi[i]; }
        sa = __builtin_amdgcn_wmma_f32_16x16x32_f16(false, ka, false, qb[c],
                                                    (short)0, sa, false, false);
      }
      s[t] = sa;
    }

    // ---------- online softmax (per-lane scalar state) ----------
    float mb = s[0][0];
    for (int t = 0; t < 4; ++t)
      for (int j = 0; j < 8; ++j) mb = fmaxf(mb, s[t][j]);
    mb = fmaxf(mb, __shfl_xor(mb, 16, 32));
    const float mnew = fmaxf(m, mb);
    const float corr = __builtin_amdgcn_exp2f(m - mnew);
    m = mnew;

    float rs = 0.f;
    for (int t = 0; t < 4; ++t) {
      v8h ph;
      for (int j = 0; j < 8; ++j) {
        float p = __builtin_amdgcn_exp2f(s[t][j] - mnew);
        rs += p;
        ph[j] = (_Float16)p;
      }
      *(v8h*)&Plds[wave][l16 * LSTR + 16 * t + 8 * half] = ph;
    }
    rs += __shfl_xor(rs, 16, 32);
    l = l * corr + rs;
    for (int t = 0; t < 4; ++t)
      for (int j = 0; j < 8; ++j) acc[t][j] *= corr;

    // ---------- acc^T += V^T * P^T ----------
    v16h pb[2];
    for (int c = 0; c < 2; ++c)
      pb[c] = *(const v16h*)&Plds[wave][l16 * LSTR + 32 * c + 16 * half];
    for (int t = 0; t < 4; ++t) {
      const int drow = t * 16 + l16;
      for (int c = 0; c < 2; ++c) {
        v8h lo = *(const v8h*)&Vc[drow * LSTR + 32 * c + 16 * half];
        v8h hi = *(const v8h*)&Vc[drow * LSTR + 32 * c + 16 * half + 8];
        v16h va;
        for (int i = 0; i < 8; ++i) { va[i] = lo[i]; va[8 + i] = hi[i]; }
        acc[t] = __builtin_amdgcn_wmma_f32_16x16x32_f16(false, va, false, pb[c],
                                                        (short)0, acc[t], false, false);
      }
    }
    __syncthreads();   // all waves done reading `cur` before it is DMA-overwritten
  }

  const float inv = 1.0f / l;
  float* optr = O + ((size_t)(b * SEQ + qrow) * CH + h * HD);
  for (int t = 0; t < 4; ++t) {
    v4f o0, o1;
    for (int i = 0; i < 4; ++i) { o0[i] = acc[t][i] * inv; o1[i] = acc[t][4 + i] * inv; }
    *(v4f*)(optr + 16 * t + 8 * half)     = o0;
    *(v4f*)(optr + 16 * t + 8 * half + 4) = o1;
  }
}

// ============================================================================
// Path B fallback (round-2 kernel): f32 K/V staged through registers, used if
// d_ws is too small for the converted copies.
// ============================================================================
__launch_bounds__(WAVES * 32)
__global__ void fa_wmma_kernel(const float* __restrict__ Q,
                               const float* __restrict__ K,
                               const float* __restrict__ V,
                               float* __restrict__ O) {
  __shared__ _Float16 Klds[KBLK * LSTR];
  __shared__ _Float16 Vlds[HD * LSTR];
  __shared__ _Float16 Plds[WAVES][QW * LSTR];

  const int bh   = blockIdx.y;
  const int b    = bh / HEADS;
  const int h    = bh % HEADS;
  const int q0   = blockIdx.x * QB;
  const int tid  = threadIdx.x;
  const int wave = tid >> 5;
  const int lane = tid & 31;
  const int l16  = lane & 15;
  const int half = lane >> 4;

  const float sl = 0.125f * 1.44269504088896f;

  const int qrow = q0 + wave * QW + l16;
  const float* qptr = Q + ((size_t)(b * SEQ + qrow) * CH + h * HD);
  v16h qb[2];
  for (int c = 0; c < 2; ++c) {
    const float* p0 = qptr + 32 * c + 16 * half;
    v4f f0 = *(const v4f*)(p0);
    v4f f1 = *(const v4f*)(p0 + 4);
    v4f f2 = *(const v4f*)(p0 + 8);
    v4f f3 = *(const v4f*)(p0 + 12);
    for (int i = 0; i < 4; ++i) {
      qb[c][i]      = (_Float16)(f0[i] * sl);
      qb[c][4 + i]  = (_Float16)(f1[i] * sl);
      qb[c][8 + i]  = (_Float16)(f2[i] * sl);
      qb[c][12 + i] = (_Float16)(f3[i] * sl);
    }
  }

  float m = -1e30f, l = 0.f;
  v8f vzero = {};
  v8f acc[4];
  for (int t = 0; t < 4; ++t) acc[t] = vzero;

  const int skey = tid >> 1;
  const int sd0  = (tid & 1) * 32;
  const float* kbase = K + ((size_t)(b * SEQ + skey) * CH + h * HD + sd0);
  const int dv  = lane * 2;
  const int kv0 = wave * 16;
  const float* vbase = V + ((size_t)(b * SEQ + kv0) * CH + h * HD + dv);

  v4f   kreg[8];
  float2 vreg[16];
  for (int i = 0; i < 8; ++i)  kreg[i] = *(const v4f*)(kbase + 4 * i);
  for (int k = 0; k < 16; ++k) vreg[k] = *(const float2*)(vbase + (size_t)k * CH);

  for (int kb = 0; kb < SEQ; kb += KBLK) {
    for (int i = 0; i < 8; ++i) {
      v4h hh;
      for (int u = 0; u < 4; ++u) hh[u] = (_Float16)kreg[i][u];
      *(v4h*)&Klds[skey * LSTR + sd0 + 4 * i] = hh;
    }
    {
      v16h c0, c1;
      for (int k = 0; k < 16; ++k) { c0[k] = (_Float16)vreg[k].x; c1[k] = (_Float16)vreg[k].y; }
      *(v16h*)&Vlds[(dv + 0) * LSTR + kv0] = c0;
      *(v16h*)&Vlds[(dv + 1) * LSTR + kv0] = c1;
    }
    __syncthreads();

    if (kb + KBLK < SEQ) {
      const float* kn = kbase + (size_t)(kb + KBLK) * CH;
      const float* vn = vbase + (size_t)(kb + KBLK) * CH;
      for (int i = 0; i < 8; ++i)  kreg[i] = *(const v4f*)(kn + 4 * i);
      for (int k = 0; k < 16; ++k) vreg[k] = *(const float2*)(vn + (size_t)k * CH);
    }

    v8f s[4];
    for (int t = 0; t < 4; ++t) {
      v8f sa = vzero;
      const int keyr = t * 16 + l16;
      for (int c = 0; c < 2; ++c) {
        v8h lo = *(const v8h*)&Klds[keyr * LSTR + 32 * c + 16 * half];
        v8h hi = *(const v8h*)&Klds[keyr * LSTR + 32 * c + 16 * half + 8];
        v16h ka;
        for (int i = 0; i < 8; ++i) { ka[i] = lo[i]; ka[8 + i] = hi[i]; }
        sa = __builtin_amdgcn_wmma_f32_16x16x32_f16(false, ka, false, qb[c],
                                                    (short)0, sa, false, false);
      }
      s[t] = sa;
    }

    float mb = s[0][0];
    for (int t = 0; t < 4; ++t)
      for (int j = 0; j < 8; ++j) mb = fmaxf(mb, s[t][j]);
    mb = fmaxf(mb, __shfl_xor(mb, 16, 32));
    const float mnew = fmaxf(m, mb);
    const float corr = __builtin_amdgcn_exp2f(m - mnew);
    m = mnew;

    float rs = 0.f;
    for (int t = 0; t < 4; ++t) {
      v8h ph;
      for (int j = 0; j < 8; ++j) {
        float p = __builtin_amdgcn_exp2f(s[t][j] - mnew);
        rs += p;
        ph[j] = (_Float16)p;
      }
      *(v8h*)&Plds[wave][l16 * LSTR + 16 * t + 8 * half] = ph;
    }
    rs += __shfl_xor(rs, 16, 32);
    l = l * corr + rs;
    for (int t = 0; t < 4; ++t)
      for (int j = 0; j < 8; ++j) acc[t][j] *= corr;

    v16h pb[2];
    for (int c = 0; c < 2; ++c)
      pb[c] = *(const v16h*)&Plds[wave][l16 * LSTR + 32 * c + 16 * half];
    for (int t = 0; t < 4; ++t) {
      const int drow = t * 16 + l16;
      for (int c = 0; c < 2; ++c) {
        v8h lo = *(const v8h*)&Vlds[drow * LSTR + 32 * c + 16 * half];
        v8h hi = *(const v8h*)&Vlds[drow * LSTR + 32 * c + 16 * half + 8];
        v16h va;
        for (int i = 0; i < 8; ++i) { va[i] = lo[i]; va[8 + i] = hi[i]; }
        acc[t] = __builtin_amdgcn_wmma_f32_16x16x32_f16(false, va, false, pb[c],
                                                        (short)0, acc[t], false, false);
      }
    }
    __syncthreads();
  }

  const float inv = 1.0f / l;
  float* optr = O + ((size_t)(b * SEQ + qrow) * CH + h * HD);
  for (int t = 0; t < 4; ++t) {
    v4f o0, o1;
    for (int i = 0; i < 4; ++i) { o0[i] = acc[t][i] * inv; o1[i] = acc[t][4 + i] * inv; }
    *(v4f*)(optr + 16 * t + 8 * half)     = o0;
    *(v4f*)(optr + 16 * t + 8 * half + 4) = o1;
  }
}

extern "C" void kernel_launch(void* const* d_in, const int* in_sizes, int n_in,
                              void* d_out, int out_size, void* d_ws, size_t ws_size,
                              hipStream_t stream) {
  const float* q = (const float*)d_in[0];
  const float* k = (const float*)d_in[1];
  const float* v = (const float*)d_in[2];
  float* o = (float*)d_out;

  dim3 grid(SEQ / QB, BATCH * HEADS);
  dim3 block(WAVES * 32);

  const size_t elems = (size_t)BATCH * HEADS * SEQ * HD;
  const size_t convBytes = 2 * elems * sizeof(_Float16);   // Kh + Vth, 8 MB

  if (ws_size >= convBytes) {
    _Float16* Kh  = (_Float16*)d_ws;
    _Float16* Vth = Kh + elems;
    convert_kv<<<dim3(SEQ / 64, BATCH * HEADS), 256, 0, stream>>>(k, v, Kh, Vth);
    fa_wmma_async<<<grid, block, 0, stream>>>(q, Kh, Vth, o);
  } else {
    fa_wmma_kernel<<<grid, block, 0, stream>>>(q, k, v, o);
  }
}
